// BEVGen_22565758174092
// MI455X (gfx1250) — matrix operand order repaired and checked by reading, not compiled, not run on hardware
//
#include <hip/hip_runtime.h>

// ---------------- problem constants (from reference) ----------------
#define BS    4
#define CIN   256
#define HF    56
#define WF    120
#define CF    64
#define DD    48          // depth bins
#define HW    (HF*WF)     // 6720
#define NPIX  (BS*HW)     // 26880
#define BEVX  128
#define BEVY  128
#define BEVHW (BEVX*BEVY) // 16384

// ---------------- WMMA fragment types (wave32, gfx1250) ----------------
typedef __attribute__((ext_vector_type(16))) __bf16 v16bf;
typedef __attribute__((ext_vector_type(8)))  float  v8f;

union FragBF { v16bf v; unsigned int u[8]; uint4 q[2]; };

__device__ __forceinline__ unsigned short f2bf(float f) {
    unsigned int u = __float_as_uint(f);
    unsigned int r = u + 0x7FFFu + ((u >> 16) & 1u);   // round-to-nearest-even
    return (unsigned short)(r >> 16);
}

// ---------------- stage 0a: state (B,CIN,H,W) f32 -> xT[(b,h,w),ci] bf16 ----------------
__global__ void prep_x_kernel(const float* __restrict__ state, unsigned short* __restrict__ xT) {
    size_t t = (size_t)blockIdx.x * blockDim.x + threadIdx.x;
    if (t >= (size_t)NPIX * CIN) return;
    int ci = (int)(t % CIN);
    size_t p = t / CIN;                 // (b,h,w) flattened
    int b  = (int)(p / HW);
    int hw = (int)(p % HW);
    float v = state[((size_t)b * CIN + ci) * HW + hw];
    xT[t] = f2bf(v);
}

// ---------------- stage 0b: w (CF,CIN,5,5) f32 -> wT[tap][co][ci] bf16 ----------------
__global__ void prep_w_kernel(const float* __restrict__ w, unsigned short* __restrict__ wT) {
    int t = blockIdx.x * blockDim.x + threadIdx.x;
    if (t >= 25 * CF * CIN) return;
    int ci   = t % CIN;
    int rest = t / CIN;
    int co   = rest % CF;
    int tap  = rest / CF;
    wT[t] = f2bf(w[((size_t)co * CIN + ci) * 25 + tap]);
}

// ---------------- stage 1: dual 5x5 conv as implicit GEMM via WMMA ----------------
// GEMM: M = NPIX pixels, N = CF = 64, K = CIN per tap, 25 taps.
// Block = 256 threads = 8 waves. Block tile = 64 pixels x 64 channels.
// wave: ni = wave>>1 -> n0 = ni*16 ; mi = wave&1 -> two 16-row M-subtiles at
//       m0 = blk*64 + mi*32 and m0+16. Each wave: 4 accumulators (featx2, decx2).
//
// Fragment layouts (ISA 7.12.2, wave32):
//  A 16x32 bf16 : lane<16 -> M=lane,   K = {0..7} in u[0..3], {16..23} in u[4..7]
//                 lane>=16 -> same M,  K offset +8 (i.e. {8..15},{24..31})
//                 => two contiguous 16B runs: base = row + ks + 8*khalf, base+16
//  B 32x16 bf16 : lane%16 = N, K same as A: {0..7}+8*khalf, {16..23}+8*khalf
//                 => base = wrow + ks + 8*khalf, base+16 (two 16B loads)
//  C 16x16 f32  : vgpr r -> M = r + 8*khalf, N = lane%16
__global__ __launch_bounds__(256) void conv_wmma_kernel(
    const unsigned short* __restrict__ xT,     // [NPIX][CIN] bf16
    const unsigned short* __restrict__ wTf,    // [25][CF][CIN] bf16 (feat)
    const unsigned short* __restrict__ wTd,    // [25][CF][CIN] bf16 (dec)
    const float* __restrict__ feat_b, const float* __restrict__ feat_g, const float* __restrict__ feat_bt,
    const float* __restrict__ dec_b,  const float* __restrict__ dec_g,  const float* __restrict__ dec_bt,
    float* __restrict__ xfeat,                 // [NPIX][CF]
    float* __restrict__ dfeat)                 // [NPIX][CF]
{
    const int wave = threadIdx.x >> 5;         // 0..7
    const int lane = threadIdx.x & 31;
    const int mi = wave & 1, ni = wave >> 1;
    const int m0 = blockIdx.x * 64 + mi * 32;  // first 16-row M-subtile of this wave
    const int n0 = ni * 16;                    // output-channel base

    const int khalf = lane >> 4;               // 0|1: K-half select
    const int lmod  = lane & 15;

    // per-lane A rows for the two M-subtiles
    int mr[2], pb[2], ph[2], pw[2];
    #pragma unroll
    for (int t = 0; t < 2; ++t) {
        mr[t] = m0 + 16 * t + lmod;
        pb[t] = mr[t] / HW;
        int hw = mr[t] % HW;
        ph[t] = hw / WF;
        pw[t] = hw % WF;
    }

    const int ncol = n0 + lmod;                // this lane's B/C column (channel)

    v8f accF0 = {0.f,0.f,0.f,0.f,0.f,0.f,0.f,0.f};
    v8f accF1 = accF0, accD0 = accF0, accD1 = accF0;

    const uint4 z4 = make_uint4(0u, 0u, 0u, 0u);

    for (int tap = 0; tap < 25; ++tap) {
        const int dh = tap / 5 - 2, dw = tap % 5 - 2;
        const unsigned short* arow[2];
        bool valid[2];
        #pragma unroll
        for (int t = 0; t < 2; ++t) {
            const int hh = ph[t] + dh, ww = pw[t] + dw;
            valid[t] = (hh >= 0) & (hh < HF) & (ww >= 0) & (ww < WF);
            arow[t]  = xT + ((size_t)((pb[t] * HF + hh) * WF + ww)) * CIN;
        }
        const unsigned short* bfrow = wTf + ((size_t)tap * CF + ncol) * CIN;
        const unsigned short* bdrow = wTd + ((size_t)tap * CF + ncol) * CIN;

        // prefetch next tap's activation row into near cache (global_prefetch_b8)
        if (tap < 24) {
            const int nh = ph[0] + (tap + 1) / 5 - 2, nw = pw[0] + (tap + 1) % 5 - 2;
            if ((nh >= 0) & (nh < HF) & (nw >= 0) & (nw < WF))
                __builtin_prefetch(xT + ((size_t)((pb[0] * HF + nh) * WF + nw)) * CIN, 0, 1);
        }

        #pragma unroll 2
        for (int ks = 0; ks < CIN; ks += 32) {
            FragBF A0, A1, Bf, Bd;
            {   // A fragments: two 16B runs each
                const unsigned short* a0 = arow[0] + ks + 8 * khalf;
                const unsigned short* a1 = arow[1] + ks + 8 * khalf;
                if (valid[0]) { A0.q[0] = *(const uint4*)a0; A0.q[1] = *(const uint4*)(a0 + 16); }
                else          { A0.q[0] = z4;                A0.q[1] = z4; }
                if (valid[1]) { A1.q[0] = *(const uint4*)a1; A1.q[1] = *(const uint4*)(a1 + 16); }
                else          { A1.q[0] = z4;                A1.q[1] = z4; }
            }
            {   // B fragments: two 16B runs, same K order as A
                const unsigned short* bf = bfrow + ks + 8 * khalf;
                const unsigned short* bd = bdrow + ks + 8 * khalf;
                Bf.q[0] = *(const uint4*)bf;  Bf.q[1] = *(const uint4*)(bf + 16);
                Bd.q[0] = *(const uint4*)bd;  Bd.q[1] = *(const uint4*)(bd + 16);
            }
            accF0 = __builtin_amdgcn_wmma_f32_16x16x32_bf16(false, A0.v, false, Bf.v, (short)0, accF0, false, false);
            accF1 = __builtin_amdgcn_wmma_f32_16x16x32_bf16(false, A1.v, false, Bf.v, (short)0, accF1, false, false);
            accD0 = __builtin_amdgcn_wmma_f32_16x16x32_bf16(false, A0.v, false, Bd.v, (short)0, accD0, false, false);
            accD1 = __builtin_amdgcn_wmma_f32_16x16x32_bf16(false, A1.v, false, Bd.v, (short)0, accD1, false, false);
        }
    }

    // epilogue: y = relu(g*(conv + b) + beta), channel-last f32 stores
    const float gF = feat_g[ncol], bF = feat_b[ncol], tF = feat_bt[ncol];
    const float gD = dec_g[ncol],  bD = dec_b[ncol],  tD = dec_bt[ncol];
    #pragma unroll
    for (int r = 0; r < 8; ++r) {
        const int ma = m0 + r + 8 * khalf;        // subtile 0
        const int mb = ma + 16;                   // subtile 1
        float yf0 = fmaxf(gF * (accF0[r] + bF) + tF, 0.f);
        float yf1 = fmaxf(gF * (accF1[r] + bF) + tF, 0.f);
        float yd0 = fmaxf(gD * (accD0[r] + bD) + tD, 0.f);
        float yd1 = fmaxf(gD * (accD1[r] + bD) + tD, 0.f);
        xfeat[(size_t)ma * CF + ncol] = yf0;
        xfeat[(size_t)mb * CF + ncol] = yf1;
        dfeat[(size_t)ma * CF + ncol] = yd0;
        dfeat[(size_t)mb * CF + ncol] = yd1;
    }
}

// ---------------- stage 2: depth 1x1 conv + softmax over D ----------------
__global__ void depth_softmax_kernel(const float* __restrict__ dfeat,   // [NPIX][CF]
                                     const float* __restrict__ dw,      // [DD][CF]
                                     const float* __restrict__ db,      // [DD]
                                     float* __restrict__ depthT)        // [NPIX][DD]
{
    int p = blockIdx.x * blockDim.x + threadIdx.x;
    if (p >= NPIX) return;
    float row[CF];
    const float4* r4 = (const float4*)(dfeat + (size_t)p * CF);
    #pragma unroll
    for (int c = 0; c < CF / 4; ++c) {
        float4 v = r4[c];
        row[4*c] = v.x; row[4*c+1] = v.y; row[4*c+2] = v.z; row[4*c+3] = v.w;
    }
    float lg[DD];
    float mx = -3.0e38f;
    #pragma unroll
    for (int d = 0; d < DD; ++d) {
        float s = db[d];
        #pragma unroll
        for (int c = 0; c < CF; ++c) s += row[c] * dw[d * CF + c];
        lg[d] = s; mx = fmaxf(mx, s);
    }
    float sum = 0.f;
    #pragma unroll
    for (int d = 0; d < DD; ++d) { float e = __expf(lg[d] - mx); lg[d] = e; sum += e; }
    const float inv = 1.f / sum;
    #pragma unroll
    for (int d = 0; d < DD; ++d) depthT[(size_t)p * DD + d] = lg[d] * inv;
}

// ---------------- stage 3: geometry -> BEV cell per (d,h,w) ----------------
__global__ void geom_kernel(const float* __restrict__ intr, const float* __restrict__ extr,
                            int* __restrict__ cell)
{
    int t = blockIdx.x * blockDim.x + threadIdx.x;
    if (t >= DD * HW) return;
    const int d = t / HW, hw = t % HW, h = hw / WF, w = hw % WF;
    // general 3x3 inverse of intr (adjugate / det), matching jnp.linalg.inv
    const float a = intr[0], bb = intr[1], c = intr[2];
    const float dk = intr[3], e = intr[4], f = intr[5];
    const float g = intr[6], hh = intr[7], i = intr[8];
    const float A = e*i - f*hh, B = -(dk*i - f*g), C = dk*hh - e*g;
    const float inv = 1.f / (a*A + bb*B + c*C);
    const float K00 = A*inv,               K01 = -(bb*i - c*hh)*inv, K02 = (bb*f - c*e)*inv;
    const float K10 = B*inv,               K11 =  (a*i  - c*g)*inv,  K12 = -(a*f - c*dk)*inv;
    const float K20 = C*inv,               K21 = -(a*hh - bb*g)*inv, K22 = (a*e - bb*dk)*inv;
    const float ds = 2.0f + (float)d;                 // D0 + d*DSTEP
    const float u  = ((float)w + 0.5f) * 8.0f * ds;   // u*ds
    const float v  = ((float)h + 0.5f) * 8.0f * ds;   // v*ds
    const float cx = K00*u + K01*v + K02*ds;
    const float cy = K10*u + K11*v + K12*ds;
    const float cz = K20*u + K21*v + K22*ds;
    const float ex = extr[0]*cx + extr[1]*cy + extr[2]*cz + extr[3];
    const float ey = extr[4]*cx + extr[5]*cy + extr[6]*cz + extr[7];
    const int ix = (int)floorf(ex / 0.6f + 32.0f + (float)(BEVX/2));
    const int iy = (int)floorf(ey / 0.6f + (float)(BEVY/2));
    const bool valid = (ix >= 0) & (ix < BEVX) & (iy >= 0) & (iy < BEVY);
    cell[t] = valid ? (iy * BEVX + ix) : -1;
}

// ---------------- stage 4: zero + lift-splat scatter ----------------
__global__ void zero_kernel(float* __restrict__ out, int n) {
    int t = blockIdx.x * blockDim.x + threadIdx.x;
    if (t < n) out[t] = 0.f;
}

__global__ void scatter_kernel(const float* __restrict__ depthT,  // [NPIX][DD]
                               const float* __restrict__ xfeat,   // [NPIX][CF]
                               const int*   __restrict__ cell,    // [DD*HW]
                               float* __restrict__ out)           // [BS][CF][BEVY][BEVX]
{
    int t = blockIdx.x * blockDim.x + threadIdx.x;   // over BS * DD * HW
    if (t >= BS * DD * HW) return;
    const int b   = t / (DD * HW);
    const int dhw = t % (DD * HW);
    const int c = cell[dhw];
    if (c < 0) return;
    const int d  = dhw / HW;
    const int hw = dhw % HW;
    const float dep = depthT[((size_t)(b * HW + hw)) * DD + d];
    const float4* xr = (const float4*)(xfeat + ((size_t)(b * HW + hw)) * CF);
    float* orow = out + (size_t)b * CF * BEVHW + c;
    #pragma unroll 4
    for (int q = 0; q < CF / 4; ++q) {
        float4 x4 = xr[q];
        atomicAdd(orow + (size_t)(4*q+0) * BEVHW, dep * x4.x);
        atomicAdd(orow + (size_t)(4*q+1) * BEVHW, dep * x4.y);
        atomicAdd(orow + (size_t)(4*q+2) * BEVHW, dep * x4.z);
        atomicAdd(orow + (size_t)(4*q+3) * BEVHW, dep * x4.w);
    }
}

// ---------------- host-side launcher ----------------
extern "C" void kernel_launch(void* const* d_in, const int* in_sizes, int n_in,
                              void* d_out, int out_size, void* d_ws, size_t ws_size,
                              hipStream_t stream) {
    const float* state   = (const float*)d_in[0];
    const float* intr    = (const float*)d_in[1];
    const float* extr    = (const float*)d_in[2];
    const float* feat_w  = (const float*)d_in[3];
    const float* feat_b  = (const float*)d_in[4];
    const float* feat_g  = (const float*)d_in[5];
    const float* feat_bt = (const float*)d_in[6];
    const float* dec_w   = (const float*)d_in[7];
    const float* dec_b   = (const float*)d_in[8];
    const float* dec_g   = (const float*)d_in[9];
    const float* dec_bt  = (const float*)d_in[10];
    const float* depth_w = (const float*)d_in[11];
    const float* depth_b = (const float*)d_in[12];
    float* out = (float*)d_out;

    // workspace layout (all 256B aligned by construction)
    char* ws = (char*)d_ws;
    size_t off = 0;
    unsigned short* xT  = (unsigned short*)(ws + off); off += (size_t)NPIX * CIN * 2;      // 13.76 MB
    unsigned short* wTf = (unsigned short*)(ws + off); off += (size_t)25 * CF * CIN * 2;   // 0.82 MB
    unsigned short* wTd = (unsigned short*)(ws + off); off += (size_t)25 * CF * CIN * 2;   // 0.82 MB
    float* xfeat  = (float*)(ws + off); off += (size_t)NPIX * CF * 4;                      // 6.88 MB
    float* dfeat  = (float*)(ws + off); off += (size_t)NPIX * CF * 4;                      // 6.88 MB
    float* depthT = (float*)(ws + off); off += (size_t)NPIX * DD * 4;                      // 5.16 MB
    int*   cell   = (int*)(ws + off);   off += (size_t)DD * HW * 4;                        // 1.29 MB

    const int T = 256;

    // stage 0: bf16 repack
    {
        size_t n = (size_t)NPIX * CIN;
        prep_x_kernel<<<(unsigned)((n + T - 1) / T), T, 0, stream>>>(state, xT);
        int nw = 25 * CF * CIN;
        prep_w_kernel<<<(nw + T - 1) / T, T, 0, stream>>>(feat_w, wTf);
        prep_w_kernel<<<(nw + T - 1) / T, T, 0, stream>>>(dec_w, wTd);
    }

    // stage 1: dual conv via WMMA (420 blocks x 8 waves; block tile 64x64)
    conv_wmma_kernel<<<NPIX / 64, 256, 0, stream>>>(
        xT, wTf, wTd,
        feat_b, feat_g, feat_bt, dec_b, dec_g, dec_bt,
        xfeat, dfeat);

    // stage 2: depth head + softmax
    depth_softmax_kernel<<<(NPIX + T - 1) / T, T, 0, stream>>>(dfeat, depth_w, depth_b, depthT);

    // stage 3: geometry (batch-independent)
    geom_kernel<<<(DD * HW + T - 1) / T, T, 0, stream>>>(intr, extr, cell);

    // stage 4: zero output then lift-splat scatter
    {
        int n = BS * CF * BEVHW;
        zero_kernel<<<(n + T - 1) / T, T, 0, stream>>>(out, n);
        int ns = BS * DD * HW;
        scatter_kernel<<<(ns + T - 1) / T, T, 0, stream>>>(depthT, xfeat, cell, out);
    }
}
